// SimpleGraphModel_68917045231738
// MI455X (gfx1250) — matrix operand order repaired and checked by reading, not compiled, not run on hardware
//
#include <hip/hip_runtime.h>

#define F_IN 128
#define HID 64
#define OUTD 300
#define OUTP 304  // padded to multiple of 16

typedef __attribute__((ext_vector_type(16))) __bf16 v16bf;
typedef __attribute__((ext_vector_type(8)))  float  v8f;
typedef __attribute__((ext_vector_type(4)))  unsigned int u32x4;
typedef __attribute__((ext_vector_type(2)))  unsigned int u32x2;

union Frag { v16bf v; u32x4 q[2]; };

__device__ __forceinline__ unsigned short f2bf(float f) {
  unsigned u = __float_as_uint(f);
  unsigned r = 0x7FFFu + ((u >> 16) & 1u);   // round-to-nearest-even
  return (unsigned short)((u + r) >> 16);
}

// Native L2-side f32 atomic add (relaxed, device scope) -> global_atomic_add_f32
__device__ __forceinline__ void atomic_fadd(float* p, float v) {
  __hip_atomic_fetch_add(p, v, __ATOMIC_RELAXED, __HIP_MEMORY_SCOPE_AGENT);
}

// ---------------------------------------------------------------------------
// K1: zero agg + cnt in workspace (harness poisons ws; must re-zero each call)
// ---------------------------------------------------------------------------
__global__ void zero_kernel(float* __restrict__ p, long n) {
  const long stride = (long)gridDim.x * blockDim.x;
  long i = (long)blockIdx.x * blockDim.x + threadIdx.x;
  const long n4 = n >> 2;
  for (long j = i; j < n4; j += stride) {
    u32x4 z = {0u, 0u, 0u, 0u};
    *(u32x4*)(p + j * 4) = z;
  }
  const long tail = n4 << 2;
  if (i < n - tail) p[tail + i] = 0.0f;
}

// ---------------------------------------------------------------------------
// K2: edge scatter-aggregate. One wave per edge: 32 lanes x float4 = 128 cols.
// x (51.2 MB) and agg (51.2 MB) both fit in the 192 MB L2 -> gathers and
// native f32 atomics resolve in L2; HBM only sees the streaming fraction.
// ---------------------------------------------------------------------------
__global__ void edge_agg_kernel(const float* __restrict__ x, const int* __restrict__ ei,
                                float* __restrict__ agg, float* __restrict__ cnt, int E) {
  const int lane = threadIdx.x & 31;
  const int wave = threadIdx.x >> 5;
  const long e = (long)blockIdx.x * 8 + wave;
  if (e >= E) return;                      // wave-uniform exit
  __builtin_prefetch(ei + e + 16384, 0, 2);  // global_prefetch_b8 (index stream)
  const int s = ei[e];
  const int d = ei[(long)E + e];
  const float4 v = *(const float4*)(x + (long)s * F_IN + lane * 4);
  float* ar = agg + (long)d * F_IN + lane * 4;
  atomic_fadd(ar + 0, v.x);
  atomic_fadd(ar + 1, v.y);
  atomic_fadd(ar + 2, v.z);
  atomic_fadd(ar + 3, v.w);
  if (lane == 0) atomic_fadd(cnt + d, 1.0f);
}

// ---------------------------------------------------------------------------
// K3: fused node kernel.
//   mean = agg/max(cnt,1); H = relu([mean|x] @ [Wl;Wr]^T + bl);
//   out = H @ Wout^T + bout.
// 8 waves/block, 1 tile of 16 nodes per wave. Weights live in LDS as bf16.
// WMMA: v_wmma_f32_16x16x32_bf16, fragments per ISA 7.12.2 wave32 layouts.
// ---------------------------------------------------------------------------
#define SMEM_BYTES (HID*F_IN*2*2 + OUTP*HID*2 + (HID+OUTP)*4 + 8*16*256*2 + 8*16*64*2)

__global__ __launch_bounds__(256, 2) void node_fused_kernel(
    const float* __restrict__ x, const float* __restrict__ agg,
    const float* __restrict__ cnt, const float* __restrict__ Wl,
    const float* __restrict__ bl, const float* __restrict__ Wr,
    const float* __restrict__ Wout, const float* __restrict__ bout,
    float* __restrict__ out, int nNodes, int nTiles)
{
  extern __shared__ char smem[];
  unsigned short* sWl = (unsigned short*)smem;            // [64][128] bf16
  unsigned short* sWr = sWl + HID * F_IN;                 // [64][128] bf16
  unsigned short* sWo = sWr + HID * F_IN;                 // [304][64] bf16 (rows>=300 zero)
  float* sBl = (float*)(sWo + OUTP * HID);                // [64]
  float* sBo = sBl + HID;                                 // [304]
  unsigned short* sA = (unsigned short*)(sBo + OUTP);     // [8][16][256] bf16 per-wave
  unsigned short* sH = sA + 8 * 16 * 256;                 // [8][16][64]  bf16 per-wave

  // ---- stage weights (block-cooperative, once) ----
  for (int i = threadIdx.x; i < HID * F_IN; i += blockDim.x) {
    sWl[i] = f2bf(Wl[i]);
    sWr[i] = f2bf(Wr[i]);
  }
  for (int i = threadIdx.x; i < OUTP * HID; i += blockDim.x) {
    int r = i / HID, c = i - r * HID;
    sWo[i] = (r < OUTD) ? f2bf(Wout[r * HID + c]) : (unsigned short)0;
  }
  for (int i = threadIdx.x; i < HID; i += blockDim.x)  sBl[i] = bl[i];
  for (int i = threadIdx.x; i < OUTP; i += blockDim.x) sBo[i] = (i < OUTD) ? bout[i] : 0.0f;
  __syncthreads();

  const int lane = threadIdx.x & 31;
  const int wave = threadIdx.x >> 5;
  const int half = lane >> 4;     // 0: lanes 0-15, 1: lanes 16-31
  const int m    = lane & 15;
  unsigned short* aW = sA + wave * 16 * 256;
  unsigned short* hW = sH + wave * 16 * 64;

  for (long tb = blockIdx.x; tb * 8 < nTiles; tb += gridDim.x) {
    const long tile = tb * 8 + wave;
    const bool valid = tile < nTiles;
    const long t0 = valid ? tile : 0;

    // ---- stage A = [mean | x] (16 x 256) bf16 into this wave's LDS slice ----
    {
      const int row = lane >> 1;            // 2 lanes per node row
      const int cb  = (lane & 1) * 64;      // half-row each
      long node = t0 * 16 + row;
      node = node < nNodes ? node : (long)(nNodes - 1);
      const float c   = cnt[node];
      const float inv = 1.0f / fmaxf(c, 1.0f);
      const float* ag = agg + node * (long)F_IN + cb;
      const float* xr = x   + node * (long)F_IN + cb;
      unsigned short* ar = aW + row * 256 + cb;
      #pragma unroll
      for (int j = 0; j < 64; j += 4) {
        float4 v = *(const float4*)(ag + j);
        u32x2 p;
        p[0] = (unsigned)f2bf(v.x * inv) | ((unsigned)f2bf(v.y * inv) << 16);
        p[1] = (unsigned)f2bf(v.z * inv) | ((unsigned)f2bf(v.w * inv) << 16);
        *(u32x2*)(ar + j) = p;
        float4 w = *(const float4*)(xr + j);
        u32x2 q;
        q[0] = (unsigned)f2bf(w.x) | ((unsigned)f2bf(w.y) << 16);
        q[1] = (unsigned)f2bf(w.z) | ((unsigned)f2bf(w.w) << 16);
        *(u32x2*)(ar + 128 + j) = q;
      }
      // prefetch next tile's rows toward L2/L0
      long tn = (tb + (long)gridDim.x) * 8 + wave;
      if (tn < nTiles) {
        long nn = tn * 16 + row;
        __builtin_prefetch(agg + nn * (long)F_IN + cb, 0, 3);
        __builtin_prefetch(x   + nn * (long)F_IN + cb, 0, 3);
      }
    }
    asm volatile("s_wait_dscnt 0x0" ::: "memory");  // wave-local LDS RAW fence

    // ---- GEMM1: H(16x64) = A(16x256) @ [Wl;Wr]^T + bl, ReLU ----
    #pragma unroll
    for (int nt = 0; nt < HID / 16; ++nt) {
      v8f acc;
      const float bv = sBl[nt * 16 + m];    // C layout: lane -> column
      #pragma unroll
      for (int r = 0; r < 8; ++r) acc[r] = bv;
      #pragma unroll
      for (int kb = 0; kb < 8; ++kb) {      // 8 x K=32 over fused K=256
        Frag a, b;
        const unsigned short* ab = aW + m * 256 + kb * 32;
        a.q[0] = *(const u32x4*)(ab + 8 * half);        // K = 8*half .. +7
        a.q[1] = *(const u32x4*)(ab + 16 + 8 * half);   // K = 16+8*half .. +7
        const unsigned short* wb =
            ((kb < 4) ? (sWl + kb * 32) : (sWr + (kb - 4) * 32))
            + (nt * 16 + m) * F_IN + 8 * half;           // B: K = 8*half..+7, 16+8*half..+7 of col
        b.q[0] = *(const u32x4*)(wb);
        b.q[1] = *(const u32x4*)(wb + 16);
        acc = __builtin_amdgcn_wmma_f32_16x16x32_bf16(
            false, a.v, false, b.v, (short)0, acc, false, false);
      }
      #pragma unroll
      for (int r = 0; r < 8; ++r)
        hW[(8 * half + r) * HID + nt * 16 + m] = f2bf(fmaxf(acc[r], 0.0f));
    }
    asm volatile("s_wait_dscnt 0x0" ::: "memory");

    // ---- GEMM2: OUT(16x300) = H(16x64) @ Wout^T + bout ----
    #pragma unroll
    for (int nt = 0; nt < OUTP / 16; ++nt) {
      v8f acc;
      const float bv = sBo[nt * 16 + m];
      #pragma unroll
      for (int r = 0; r < 8; ++r) acc[r] = bv;
      #pragma unroll
      for (int kb = 0; kb < 2; ++kb) {
        Frag a, b;
        const unsigned short* ab = hW + m * HID + kb * 32;
        a.q[0] = *(const u32x4*)(ab + 8 * half);
        a.q[1] = *(const u32x4*)(ab + 16 + 8 * half);
        const unsigned short* wb = sWo + (nt * 16 + m) * HID + kb * 32 + 8 * half;
        b.q[0] = *(const u32x4*)(wb);
        b.q[1] = *(const u32x4*)(wb + 16);
        acc = __builtin_amdgcn_wmma_f32_16x16x32_bf16(
            false, a.v, false, b.v, (short)0, acc, false, false);
      }
      const int col = nt * 16 + m;
      if (valid && col < OUTD) {
        #pragma unroll
        for (int r = 0; r < 8; ++r) {
          const long nr = t0 * 16 + 8 * half + r;
          if (nr < nNodes) out[nr * (long)OUTD + col] = acc[r];
        }
      }
    }
    asm volatile("s_wait_dscnt 0x0" ::: "memory");  // WAR fence before A reuse
  }
}

// ---------------------------------------------------------------------------
extern "C" void kernel_launch(void* const* d_in, const int* in_sizes, int n_in,
                              void* d_out, int out_size, void* d_ws, size_t ws_size,
                              hipStream_t stream) {
  const float* x  = (const float*)d_in[0];
  const int*   ei = (const int*)d_in[1];      // edge_index [2, E] (int32 per harness)
  const float* Wl = (const float*)d_in[2];
  const float* bl = (const float*)d_in[3];
  const float* Wr = (const float*)d_in[4];
  const float* Wo = (const float*)d_in[5];
  const float* bo = (const float*)d_in[6];
  float* out = (float*)d_out;

  const int nNodes = in_sizes[0] / F_IN;
  const int E      = in_sizes[1] / 2;

  float* agg = (float*)d_ws;                        // [nNodes][128]
  float* cnt = agg + (long)nNodes * F_IN;           // [nNodes]
  const long zn = (long)nNodes * F_IN + nNodes;

  zero_kernel<<<2048, 256, 0, stream>>>(agg, zn);
  edge_agg_kernel<<<(E + 7) / 8, 256, 0, stream>>>(x, ei, agg, cnt, E);

  const int nTiles = (nNodes + 15) / 16;
  node_fused_kernel<<<256, 256, SMEM_BYTES, stream>>>(
      x, agg, cnt, Wl, bl, Wr, Wo, bo, out, nNodes, nTiles);
}
